// EmbGCN_40286793236890
// MI455X (gfx1250) — matrix operand order repaired and checked by reading, not compiled, not run on hardware
//
#include <hip/hip_runtime.h>

// ---------------------------------------------------------------------------
// EmbGCN on MI455X (gfx1250, wave32, WMMA, TDM).
// Dominant: 68.7 GFLOP aggregation A_dyn[4096x4096] @ X[4096x2048] -> bf16 WMMA,
//   double-buffered: A tile DMA'd by the Tensor Data Mover and X tile staged
//   fragment-ready in LDS for the NEXT K-step while WMMAs consume the current.
// Per-node adaptive conv + static linear -> fused f32 WMMA (16x16x4).
// ---------------------------------------------------------------------------

typedef __bf16 bf16_t;
typedef bf16_t v16bf __attribute__((ext_vector_type(16)));
typedef float  v8f   __attribute__((ext_vector_type(8)));
typedef float  v2f   __attribute__((ext_vector_type(2)));
typedef float  f32x4 __attribute__((ext_vector_type(4)));
typedef unsigned int u32;
typedef u32    u32x4 __attribute__((ext_vector_type(4)));
typedef int    i32x4 __attribute__((ext_vector_type(4)));
typedef int    i32x8 __attribute__((ext_vector_type(8)));

#define NN   4096   // nodes
#define CIN  64
#define COUT 64
#define NB   32     // batch
#define DEMB 16
#define BJ   2048   // NB * CIN columns of the aggregation GEMM

// ---- Kernel 1: A_dyn = softmax(relu(E E^T)) row-wise, bf16 out ------------
__global__ __launch_bounds__(256) void k_embsoftmax(const float* __restrict__ E,
                                                    bf16_t* __restrict__ Adyn) {
  const int row0 = blockIdx.x * 16;
  const int tid  = threadIdx.x;
  const int lane = tid & 31;
  const int wave = tid >> 5;

  __shared__ float sE[16 * 16];
  __shared__ float sred[16 * 8];
  __shared__ float srmax[16];
  __shared__ float srinv[16];

  sE[tid & 255] = E[(row0 + (tid >> 4)) * DEMB + (tid & 15)];
  __syncthreads();

  float rmax[16];
#pragma unroll
  for (int r = 0; r < 16; ++r) rmax[r] = 0.f;  // relu output >= 0

  for (int j = 0; j < 16; ++j) {               // pass 1: row maxima
    const int c = tid + j * 256;
    float ec[DEMB];
#pragma unroll
    for (int d = 0; d < DEMB; ++d) ec[d] = E[c * DEMB + d];
#pragma unroll
    for (int r = 0; r < 16; ++r) {
      float s = 0.f;
#pragma unroll
      for (int d = 0; d < DEMB; ++d) s += sE[r * DEMB + d] * ec[d];
      s = fmaxf(s, 0.f);
      rmax[r] = fmaxf(rmax[r], s);
    }
  }
#pragma unroll
  for (int r = 0; r < 16; ++r) {
    float m = rmax[r];
    for (int off = 16; off >= 1; off >>= 1) m = fmaxf(m, __shfl_xor(m, off, 32));
    if (lane == 0) sred[r * 8 + wave] = m;
  }
  __syncthreads();
  if (tid < 16) {
    float m = sred[tid * 8];
#pragma unroll
    for (int w = 1; w < 8; ++w) m = fmaxf(m, sred[tid * 8 + w]);
    srmax[tid] = m;
  }
  __syncthreads();

  float rsum[16];                              // pass 2: sum of exp
#pragma unroll
  for (int r = 0; r < 16; ++r) rsum[r] = 0.f;
  for (int j = 0; j < 16; ++j) {
    const int c = tid + j * 256;
    float ec[DEMB];
#pragma unroll
    for (int d = 0; d < DEMB; ++d) ec[d] = E[c * DEMB + d];
#pragma unroll
    for (int r = 0; r < 16; ++r) {
      float s = 0.f;
#pragma unroll
      for (int d = 0; d < DEMB; ++d) s += sE[r * DEMB + d] * ec[d];
      s = fmaxf(s, 0.f);
      rsum[r] += __expf(s - srmax[r]);
    }
  }
#pragma unroll
  for (int r = 0; r < 16; ++r) {
    float m = rsum[r];
    for (int off = 16; off >= 1; off >>= 1) m += __shfl_xor(m, off, 32);
    if (lane == 0) sred[r * 8 + wave] = m;
  }
  __syncthreads();
  if (tid < 16) {
    float m = 0.f;
#pragma unroll
    for (int w = 0; w < 8; ++w) m += sred[tid * 8 + w];
    srinv[tid] = 1.f / m;
  }
  __syncthreads();

  for (int j = 0; j < 16; ++j) {               // pass 3: write bf16
    const int c = tid + j * 256;
    float ec[DEMB];
#pragma unroll
    for (int d = 0; d < DEMB; ++d) ec[d] = E[c * DEMB + d];
#pragma unroll
    for (int r = 0; r < 16; ++r) {
      float s = 0.f;
#pragma unroll
      for (int d = 0; d < DEMB; ++d) s += sE[r * DEMB + d] * ec[d];
      s = fmaxf(s, 0.f);
      float w = __expf(s - srmax[r]) * srinv[r];
      Adyn[(size_t)(row0 + r) * NN + c] = (bf16_t)w;
    }
  }
}

// ---- row sums of (adj + 0.5 I) -> sd = rsqrt(rowsum) ----------------------
__global__ __launch_bounds__(256) void k_rowsum(const float* __restrict__ adj,
                                                float* __restrict__ sd) {
  const int n   = blockIdx.x;
  const int tid = threadIdx.x;
  float s = 0.f;
#pragma unroll
  for (int j = 0; j < 16; ++j) s += adj[(size_t)n * NN + tid + j * 256];
  for (int off = 16; off >= 1; off >>= 1) s += __shfl_xor(s, off, 32);
  __shared__ float sr[8];
  if ((tid & 31) == 0) sr[tid >> 5] = s;
  __syncthreads();
  if (tid == 0) {
    float t = 0.f;
#pragma unroll
    for (int w = 0; w < 8; ++w) t += sr[w];
    sd[n] = rsqrtf(t + 0.5f);
  }
}

__global__ __launch_bounds__(256) void k_zero(float* __restrict__ p, int n) {
  int i = blockIdx.x * 256 + threadIdx.x;
  if (i < n) p[i] = 0.f;
}

// ---- scale[m] = sum_n softmax(softmax(symnorm(adj)))[n,m] -----------------
__global__ __launch_bounds__(256) void k_static_scale(const float* __restrict__ adj,
                                                      const float* __restrict__ sd,
                                                      float* __restrict__ scale) {
  const int row0 = blockIdx.x * 16;
  const int tid  = threadIdx.x;
  const int lane = tid & 31;
  const int wave = tid >> 5;

  __shared__ float ssd[16];
  __shared__ float sred[16 * 8];
  __shared__ float smax1[16];
  __shared__ float sinv1[16];
  __shared__ float sinv2[16];

  if (tid < 16) ssd[tid] = sd[row0 + tid];
  __syncthreads();

  float rv[16];
#pragma unroll
  for (int r = 0; r < 16; ++r) rv[r] = 0.f;    // pass 1: max1 (S >= 0)
  for (int j = 0; j < 16; ++j) {
    const int c = tid + j * 256;
    const float sdc = sd[c];
#pragma unroll
    for (int r = 0; r < 16; ++r) {
      float a = adj[(size_t)(row0 + r) * NN + c];
      if (c == row0 + r) a += 0.5f;
      rv[r] = fmaxf(rv[r], ssd[r] * a * sdc);
    }
  }
#pragma unroll
  for (int r = 0; r < 16; ++r) {
    float m = rv[r];
    for (int off = 16; off >= 1; off >>= 1) m = fmaxf(m, __shfl_xor(m, off, 32));
    if (lane == 0) sred[r * 8 + wave] = m;
  }
  __syncthreads();
  if (tid < 16) {
    float m = sred[tid * 8];
#pragma unroll
    for (int w = 1; w < 8; ++w) m = fmaxf(m, sred[tid * 8 + w]);
    smax1[tid] = m;
  }
  __syncthreads();

#pragma unroll
  for (int r = 0; r < 16; ++r) rv[r] = 0.f;    // pass 2: sum1
  for (int j = 0; j < 16; ++j) {
    const int c = tid + j * 256;
    const float sdc = sd[c];
#pragma unroll
    for (int r = 0; r < 16; ++r) {
      float a = adj[(size_t)(row0 + r) * NN + c];
      if (c == row0 + r) a += 0.5f;
      rv[r] += __expf(ssd[r] * a * sdc - smax1[r]);
    }
  }
#pragma unroll
  for (int r = 0; r < 16; ++r) {
    float m = rv[r];
    for (int off = 16; off >= 1; off >>= 1) m += __shfl_xor(m, off, 32);
    if (lane == 0) sred[r * 8 + wave] = m;
  }
  __syncthreads();
  if (tid < 16) {
    float m = 0.f;
#pragma unroll
    for (int w = 0; w < 8; ++w) m += sred[tid * 8 + w];
    sinv1[tid] = 1.f / m;                      // == maxP of row
  }
  __syncthreads();

#pragma unroll
  for (int r = 0; r < 16; ++r) rv[r] = 0.f;    // pass 3: sum2 of exp(P-maxP)
  for (int j = 0; j < 16; ++j) {
    const int c = tid + j * 256;
    const float sdc = sd[c];
#pragma unroll
    for (int r = 0; r < 16; ++r) {
      float a = adj[(size_t)(row0 + r) * NN + c];
      if (c == row0 + r) a += 0.5f;
      float P = __expf(ssd[r] * a * sdc - smax1[r]) * sinv1[r];
      rv[r] += __expf(P - sinv1[r]);
    }
  }
#pragma unroll
  for (int r = 0; r < 16; ++r) {
    float m = rv[r];
    for (int off = 16; off >= 1; off >>= 1) m += __shfl_xor(m, off, 32);
    if (lane == 0) sred[r * 8 + wave] = m;
  }
  __syncthreads();
  if (tid < 16) {
    float m = 0.f;
#pragma unroll
    for (int w = 0; w < 8; ++w) m += sred[tid * 8 + w];
    sinv2[tid] = 1.f / m;
  }
  __syncthreads();

  for (int j = 0; j < 16; ++j) {               // pass 4: Q column partials
    const int c = tid + j * 256;
    const float sdc = sd[c];
    float qs = 0.f;
#pragma unroll
    for (int r = 0; r < 16; ++r) {
      float a = adj[(size_t)(row0 + r) * NN + c];
      if (c == row0 + r) a += 0.5f;
      float P = __expf(ssd[r] * a * sdc - smax1[r]) * sinv1[r];
      qs += __expf(P - sinv1[r]) * sinv2[r];
    }
    atomicAdd(&scale[c], qs);
  }
}

// ---- Kernel 2 (dominant): agg[n, b*64+c] = sum_m Adyn[n,m] X[m, b*64+c] ---
// bf16 WMMA 16x16x32.  WG tile 128x128, wave tile 32x64.
// Double-buffered pipeline: TDM loads next A tile + threads stage next X tile
// while WMMAs consume the current buffers; s_wait_tensorcnt 0 + one barrier
// per K-step.
__global__ __launch_bounds__(256) void k_aggregate(const bf16_t* __restrict__ Adyn,
                                                   const float* __restrict__ x,
                                                   float* __restrict__ agg) {
  __shared__ __align__(16) bf16_t sA[2][128 * 32];  // 2 x 8KB, linear [row][k]
  __shared__ v16bf sXf[2][8 * 32];                  // 2 x 8KB, frag-ready

  const int tid  = threadIdx.x;
  const int lane = tid & 31;
  const int wave = tid >> 5;
  const int j0   = blockIdx.x * 128;
  const int n0   = blockIdx.y * 128;
  const int mt2  = wave >> 1;   // 0..3 : 32-row strip
  const int jt   = wave & 1;    // 0..1 : 64-col strip

  // ---- helpers --------------------------------------------------------
  auto issueA = [&](int k0, int buf) {
#if __has_builtin(__builtin_amdgcn_tensor_load_to_lds) && \
    __has_builtin(__builtin_amdgcn_s_wait_tensorcnt)
    if (wave == 0) {
      const unsigned long long ga =
          (unsigned long long)(uintptr_t)(Adyn + (size_t)n0 * NN + k0);
      const u32 lds_off = (u32)(uintptr_t)&sA[buf][0];  // low 32b = LDS offset
      u32x4 g0;
      g0[0] = 1u;                                   // count=1, user descriptor
      g0[1] = lds_off;                              // lds_addr
      g0[2] = (u32)(ga & 0xffffffffu);              // global_addr[31:0]
      g0[3] = (u32)((ga >> 32) & 0x1ffffffu) | (2u << 30);  // [56:32], type=2
      i32x8 g1;
      g1[0] = 0x00010000;        // workgroup_mask=0, data_size=2B
      g1[1] = 0x10000000;        // tensor_dim0 = 4096 (bits 63:48)
      g1[2] = 0x10000000;        // tensor_dim1 = 4096 (bits 95:80)
      g1[3] = 32 << 16;          // tile_dim0 = 32 (bits 127:112)
      g1[4] = 128;               // tile_dim1 = 128
      g1[5] = 4096;              // tensor_dim0_stride = 4096 elements
      g1[6] = 0;
      g1[7] = 0;
      i32x4 gz;
      gz[0] = 0; gz[1] = 0; gz[2] = 0; gz[3] = 0;  // 2-D: groups 2/3 unused
      i32x8 gz8;
#pragma unroll
      for (int zz = 0; zz < 8; ++zz) gz8[zz] = 0;
      __builtin_amdgcn_tensor_load_to_lds(g0, g1, gz, gz, gz8, 0);
    }
#else
#pragma unroll
    for (int s = 0; s < 2; ++s) {
      const int q  = tid + s * 256;       // 512 chunks of 8 bf16
      const int r  = q >> 2;
      const int cb = (q & 3) * 8;
      *(u32x4*)&sA[buf][r * 32 + cb] =
          *(const u32x4*)&Adyn[(size_t)(n0 + r) * NN + k0 + cb];
    }
#endif
  };

  auto stageX = [&](int k0, int buf) {
    // slot (ct,lane,h): 8 elems K = h*16+ei + 8*(lane>=16), col = ct*16+lane%16
#pragma unroll
    for (int s = 0; s < 2; ++s) {
      const int hs = tid + s * 256;       // 512 half-slots
      const int ct = hs >> 6;
      const int ln = (hs >> 1) & 31;
      const int h  = hs & 1;
      const int jj = j0 + ct * 16 + (ln & 15);
      const int b  = jj >> 6;
      const int cc = jj & 63;
      const int m0 = k0 + h * 16 + ((ln >= 16) ? 8 : 0);
      u32x4 pk;
#pragma unroll
      for (int ei = 0; ei < 4; ++ei) {
        union { bf16_t hh[2]; u32 w; } pw;
        pw.hh[0] = (bf16_t)x[((size_t)b * NN + m0 + 2 * ei) * CIN + cc];
        pw.hh[1] = (bf16_t)x[((size_t)b * NN + m0 + 2 * ei + 1) * CIN + cc];
        pk[ei] = pw.w;
      }
      ((u32x4*)&sXf[buf][ct * 32 + ln])[h] = pk;
    }
  };

  v8f acc[2][4];
#pragma unroll
  for (int i = 0; i < 2; ++i)
#pragma unroll
    for (int f = 0; f < 4; ++f)
#pragma unroll
      for (int q = 0; q < 8; ++q) acc[i][f][q] = 0.f;

  // ---- prologue: fill buffer 0 ----
  issueA(0, 0);
  stageX(0, 0);
#if __has_builtin(__builtin_amdgcn_tensor_load_to_lds) && \
    __has_builtin(__builtin_amdgcn_s_wait_tensorcnt)
  __builtin_amdgcn_s_wait_tensorcnt(0);
#endif
  __syncthreads();

  const int a0 = ((mt2 * 2 + 0) * 16 + (lane & 15)) * 32 + ((lane >= 16) ? 8 : 0);
  const int a1base =
      ((mt2 * 2 + 1) * 16 + (lane & 15)) * 32 + ((lane >= 16) ? 8 : 0);

  for (int it = 0; it < NN / 32; ++it) {
    const int cur = it & 1;
    const int nxt = cur ^ 1;
    const bool more = (it + 1) < (NN / 32);

    // ---- prefetch next tile (TDM + global->LDS staging), overlaps WMMA ----
    if (more) {
      issueA((it + 1) * 32, nxt);
      stageX((it + 1) * 32, nxt);
    }

    // ---- compute current tile: all frags up-front, then 8 WMMAs ----
    union { u32x4 q[2]; v16bf v; } fu0, fu1;
    fu0.q[0] = *(const u32x4*)&sA[cur][a0];
    fu0.q[1] = *(const u32x4*)&sA[cur][a0 + 16];
    fu1.q[0] = *(const u32x4*)&sA[cur][a1base];
    fu1.q[1] = *(const u32x4*)&sA[cur][a1base + 16];
    const v16bf af0 = fu0.v, af1 = fu1.v;
    const v16bf bf0 = sXf[cur][(jt * 4 + 0) * 32 + lane];
    const v16bf bf1 = sXf[cur][(jt * 4 + 1) * 32 + lane];
    const v16bf bf2 = sXf[cur][(jt * 4 + 2) * 32 + lane];
    const v16bf bf3 = sXf[cur][(jt * 4 + 3) * 32 + lane];

    acc[0][0] = __builtin_amdgcn_wmma_f32_16x16x32_bf16(
        false, af0, false, bf0, (short)0, acc[0][0], false, false);
    acc[1][0] = __builtin_amdgcn_wmma_f32_16x16x32_bf16(
        false, af1, false, bf0, (short)0, acc[1][0], false, false);
    acc[0][1] = __builtin_amdgcn_wmma_f32_16x16x32_bf16(
        false, af0, false, bf1, (short)0, acc[0][1], false, false);
    acc[1][1] = __builtin_amdgcn_wmma_f32_16x16x32_bf16(
        false, af1, false, bf1, (short)0, acc[1][1], false, false);
    acc[0][2] = __builtin_amdgcn_wmma_f32_16x16x32_bf16(
        false, af0, false, bf2, (short)0, acc[0][2], false, false);
    acc[1][2] = __builtin_amdgcn_wmma_f32_16x16x32_bf16(
        false, af1, false, bf2, (short)0, acc[1][2], false, false);
    acc[0][3] = __builtin_amdgcn_wmma_f32_16x16x32_bf16(
        false, af0, false, bf3, (short)0, acc[0][3], false, false);
    acc[1][3] = __builtin_amdgcn_wmma_f32_16x16x32_bf16(
        false, af1, false, bf3, (short)0, acc[1][3], false, false);

    // ---- retire next-tile DMA, publish staging, swap buffers ----
#if __has_builtin(__builtin_amdgcn_tensor_load_to_lds) && \
    __has_builtin(__builtin_amdgcn_s_wait_tensorcnt)
    __builtin_amdgcn_s_wait_tensorcnt(0);
#endif
    __syncthreads();
  }

  // epilogue: D row = q + 8*(lane>=16), col = lane%16
  const int hi8  = (lane >= 16) ? 8 : 0;
  const int colL = lane & 15;
#pragma unroll
  for (int i = 0; i < 2; ++i)
#pragma unroll
    for (int f = 0; f < 4; ++f) {
      const int j = j0 + jt * 64 + f * 16 + colL;
#pragma unroll
      for (int q = 0; q < 8; ++q) {
        const int n = n0 + (mt2 * 2 + i) * 16 + q + hi8;
        agg[(size_t)n * BJ + j] = acc[i][f][q];
      }
    }
}

// ---- Kernel 3: per-node fused adaptive conv + static gate + output --------
// sWt = W_n^T [o][K] (stride 130: bank-conflict-free, 8B aligned pairs)
// gemm1: xg[32x128] @ W_n[128x64]   (f32 WMMA 16x16x4)
// gemm2: x_n[32x64] @ lin_w^T[64x64]
#define LD3 130
__global__ __launch_bounds__(256) void k_adaptive_out(
    const float* __restrict__ x, const float* __restrict__ E,
    const float* __restrict__ Wp, const float* __restrict__ bias_pool,
    const float* __restrict__ lin_w, const float* __restrict__ lin_b,
    const float* __restrict__ agg, const float* __restrict__ scale,
    float* __restrict__ out) {
  __shared__ float sWt[COUT * LD3];   // ~32.5 KB  W_n transposed [o][ki]
  __shared__ float sXg[NB * LD3];     // ~16.3 KB  [batch][2*Cin]
  __shared__ float sE[DEMB];

  const int n   = blockIdx.x;
  const int tid = threadIdx.x;
  if (tid < DEMB) sE[tid] = E[n * DEMB + tid];
  __syncthreads();

  // build W_n^T: coalesced float4 reads of Wp (512KB total -> L2 resident)
#pragma unroll
  for (int g = 0; g < 8; ++g) {
    const int e4 = (tid + g * 256) * 4;          // 0..8188, 4 consecutive
    f32x4 a; a[0] = 0.f; a[1] = 0.f; a[2] = 0.f; a[3] = 0.f;
#pragma unroll
    for (int d = 0; d < DEMB; ++d) {
      f32x4 w = *(const f32x4*)&Wp[d * 8192 + e4];
      a = a + w * sE[d];
    }
    const int ki = e4 >> 6;
    const int o  = e4 & 63;
#pragma unroll
    for (int t4 = 0; t4 < 4; ++t4) sWt[(o + t4) * LD3 + ki] = a[t4];
  }
  // stage xg: k=0 -> x, k=1 -> agg
#pragma unroll
  for (int j = 0; j < 16; ++j) {
    const int idx = tid + j * 256;
    const int b   = idx >> 7;
    const int ki  = idx & 127;
    float v = (ki < CIN) ? x[((size_t)b * NN + n) * CIN + ki]
                         : agg[(size_t)n * BJ + b * CIN + (ki - CIN)];
    sXg[b * LD3 + ki] = v;
  }
  __syncthreads();

  const int lane = tid & 31;
  const int wave = tid >> 5;
  const int mt   = wave & 1;   // batch strip 0..1
  const int nt   = wave >> 1;  // out-col strip 0..3
  const int hi2  = (lane >= 16) ? 2 : 0;
  const int rowb = mt * 16 + (lane & 15);   // A row (batch)
  const int colo = nt * 16 + (lane & 15);   // B/D col (out channel)

  v8f a1, a2;
#pragma unroll
  for (int q = 0; q < 8; ++q) { a1[q] = 0.f; a2[q] = 0.f; }

  // gemm1: K = 128
#pragma unroll
  for (int ks = 0; ks < 32; ++ks) {
    const int kb = ks * 4 + hi2;
    v2f av = *(const v2f*)&sXg[rowb * LD3 + kb];
    v2f bv = *(const v2f*)&sWt[colo * LD3 + kb];
    a1 = __builtin_amdgcn_wmma_f32_16x16x4_f32(
        false, av, false, bv, (short)0, a1, false, false);
  }
  // gemm2: K = 64, B = lin_w^T (16KB, cache-resident; b64 loads)
#pragma unroll
  for (int ks = 0; ks < 16; ++ks) {
    const int kb = ks * 4 + hi2;
    v2f av = *(const v2f*)&sXg[rowb * LD3 + kb];
    v2f bv = *(const v2f*)&lin_w[colo * CIN + kb];
    a2 = __builtin_amdgcn_wmma_f32_16x16x4_f32(
        false, av, false, bv, (short)0, a2, false, false);
  }

  // epilogue
  float bias_no = 0.f;
#pragma unroll
  for (int d = 0; d < DEMB; ++d) bias_no += sE[d] * bias_pool[d * COUT + colo];
  const float sc = scale[n];
  const float lb = lin_b[colo];
  const int hi8  = (lane >= 16) ? 8 : 0;
#pragma unroll
  for (int q = 0; q < 8; ++q) {
    const int b = mt * 16 + q + hi8;
    float g  = a1[q] + bias_no;
    float xs = sc * a2[q] + lb;
    float sg = 1.f / (1.f + __expf(-xs));
    out[((size_t)b * NN + n) * COUT + colo] = g + sg * xs;
  }
}

// ---------------------------------------------------------------------------
extern "C" void kernel_launch(void* const* d_in, const int* in_sizes, int n_in,
                              void* d_out, int out_size, void* d_ws, size_t ws_size,
                              hipStream_t stream) {
  (void)in_sizes; (void)n_in; (void)out_size; (void)ws_size;
  const float* x         = (const float*)d_in[0];
  const float* E         = (const float*)d_in[1];
  const float* Wp        = (const float*)d_in[2];
  const float* bias_pool = (const float*)d_in[3];
  const float* lin_w     = (const float*)d_in[4];
  const float* lin_b     = (const float*)d_in[5];
  const float* adj       = (const float*)d_in[6];
  float* out = (float*)d_out;

  char* ws = (char*)d_ws;
  bf16_t* Adyn = (bf16_t*)ws;                                      // 33.5 MB
  float*  agg  = (float*)(ws + (size_t)NN * NN * sizeof(bf16_t));  // 33.5 MB
  float*  sd   = (float*)(ws + (size_t)NN * NN * sizeof(bf16_t)
                             + (size_t)NB * NN * CIN * sizeof(float));
  float*  scale = sd + NN;

  k_embsoftmax<<<NN / 16, 256, 0, stream>>>(E, Adyn);
  k_rowsum<<<NN, 256, 0, stream>>>(adj, sd);
  k_zero<<<NN / 256, 256, 0, stream>>>(scale, NN);
  k_static_scale<<<NN / 16, 256, 0, stream>>>(adj, sd, scale);
  k_aggregate<<<dim3(BJ / 128, NN / 128), 256, 0, stream>>>(Adyn, x, agg);
  k_adaptive_out<<<NN, 256, 0, stream>>>(x, E, Wp, bias_pool, lin_w, lin_b,
                                         agg, scale, out);
}
